// Puzzler_1202590843027
// MI455X (gfx1250) — matrix operand order repaired and checked by reading, not compiled, not run on hardware
//
#include <hip/hip_runtime.h>

// ---------------------------------------------------------------------------
// Problem constants (match reference)
// ---------------------------------------------------------------------------
#define BATCH 32
#define CHANS 3
#define IMH   224
#define IMW   224
#define NS    64          // K*K sample points
#define HID   256
#define MID   256
#define NCLS  1000
#define KPAD  480         // 451 padded to multiple of 32 (15 k-steps)
#define IN_DIM 451
#define OUT_DIM 259
#define NW2   32          // waves in kernel 2 (1024 threads)

static __device__ __forceinline__ float sigmoidf_(float x) { return 1.0f / (1.0f + expf(-x)); }

#define INV2S2 (1.0f / (2.0f * 5.0f * 5.0f))   // 1/(2*std^2), std = 5
#define EPSV   1e-6f

typedef __attribute__((ext_vector_type(16))) _Float16 v16h;
typedef __attribute__((ext_vector_type(8)))  _Float16 v8h;
typedef __attribute__((ext_vector_type(2)))  _Float16 v2h;
typedef __attribute__((ext_vector_type(8)))  float    v8f;

// ---------------------------------------------------------------------------
// WMMA fragment helpers (16-bit, k order used by this file, wave32)
//   A 16x32 (MxK): lane m = lane&15;  K(e) = e + (lane>=16?16:0)
//     -> per lane: one contiguous 16-half run at k = kh (kh = 0 or 16)
//   B 32x16 (KxN): lane n = lane&15;  K(e) = e + (lane>=16?16:0)
//   C 16x16:       N = lane&15;       M(r) = r + (lane>=16?8:0)
// ---------------------------------------------------------------------------
static __device__ __forceinline__ v16h ldA_lds(const _Float16* tile, int ld, int lane) {
    const int m  = lane & 15;
    const int kh = (lane >> 4) << 4;          // 0 or 16
    const _Float16* p = tile + m * ld + kh;   // 16B aligned (ld, kh multiples of 8)
    v8h lo = *(const v8h*)(p);                // k = kh .. kh+7        (ds_load_b128)
    v8h hi = *(const v8h*)(p + 8);            // k = kh+8 .. kh+15     (ds_load_b128)
    return __builtin_shufflevector(lo, hi, 0, 1, 2, 3, 4, 5, 6, 7,
                                           8, 9, 10, 11, 12, 13, 14, 15);
}

// B fragment straight from a global row-major weight matrix W[N][Kact]:
// B[k][n] = W[n][k]; each lane reads 16 contiguous floats (128b loads if aligned).
static __device__ __forceinline__ v16h ldB_wt(const float* __restrict__ W,
                                              int Kact, int N, int n0, int k0, int lane) {
    const int n  = n0 + (lane & 15);
    const int kh = (lane >> 4) << 4;          // 0 or 16
    v16h r;
    if (n < N) {
        const float* row = W + (size_t)n * Kact + (size_t)(k0 + kh);
        if ((Kact & 3) == 0) {                // rows 16B aligned -> 4x b128
            const float4* q = (const float4*)row;
            float4 f0 = q[0], f1 = q[1], f2 = q[2], f3 = q[3];
            r[0]=(_Float16)f0.x; r[1]=(_Float16)f0.y; r[2]=(_Float16)f0.z; r[3]=(_Float16)f0.w;
            r[4]=(_Float16)f1.x; r[5]=(_Float16)f1.y; r[6]=(_Float16)f1.z; r[7]=(_Float16)f1.w;
            r[8]=(_Float16)f2.x; r[9]=(_Float16)f2.y; r[10]=(_Float16)f2.z; r[11]=(_Float16)f2.w;
            r[12]=(_Float16)f3.x; r[13]=(_Float16)f3.y; r[14]=(_Float16)f3.z; r[15]=(_Float16)f3.w;
        } else {
#pragma unroll
            for (int e = 0; e < 16; ++e) {
                int k = k0 + kh + e;
                r[e] = (k < Kact) ? (_Float16)row[e] : (_Float16)0.0f;
            }
        }
    } else {
#pragma unroll
        for (int e = 0; e < 16; ++e) r[e] = (_Float16)0.0f;
    }
    return r;
}

// ---------------------------------------------------------------------------
// Kernel 1: separable Gaussian interpolation.
// One block per (batch, channel). T = wy(64x224) @ img_c(224x224) via WMMA.
// The image K-slab is staged into LDS *in B-fragment order*, so each compute
// lane fetches its whole fragment with one aligned 32B LDS load.
// Epilogue folds the wx weighting + w-reduction (shuffle tree + exactly two
// LDS float atomics per slot -> order-commutative -> deterministic).
// ---------------------------------------------------------------------------
__global__ __launch_bounds__(256) void k_gauss(const float* __restrict__ img,
                                               const float* __restrict__ loc,
                                               const float* __restrict__ scl,
                                               float* __restrict__ pix) {
    const int b = blockIdx.x / CHANS;
    const int c = blockIdx.x % CHANS;

    // fragment-ordered slab: [nt(14)][lane(32)][e(16)] halves  (14336 B)
    __shared__ alignas(16) _Float16 fragB[14 * 32 * 16];
    __shared__ float ys[NS], xs[NS], sy[NS], sx[NS], num[NS];

    const int tid  = threadIdx.x;
    const int wid  = tid >> 5;
    const int lane = tid & 31;

    const float l0 = loc[b * 2 + 0];
    const float l1 = loc[b * 2 + 1];
    const float sc = scl[b];

    if (tid < NS) {
        int i = tid >> 3, j = tid & 7;
        ys[tid]  = (((float)i * 0.125f) - l0) * sc * (float)IMH;
        xs[tid]  = (((float)j * 0.125f) - l1) * sc * (float)IMW;
        num[tid] = 0.0f;
    }
    __syncthreads();

    // fp32 denominator sums (match reference precision)
    if (tid < NS) {
        float a = 0.0f, y = ys[tid];
        for (int h = 0; h < IMH; ++h) { float d = (float)h - y; a += expf(-d * d * INV2S2); }
        sy[tid] = a;
    } else if (tid < 2 * NS) {
        int t = tid - NS;
        float a = 0.0f, x = xs[t];
        for (int w = 0; w < IMW; ++w) { float d = (float)w - x; a += expf(-d * d * INV2S2); }
        sx[t] = a;
    }

    // tile ownership: wave -> (M-tile mt, 7 N-tiles starting at nt0)
    const int mt  = wid & 3;                       // 4 M-tiles cover 64 samples
    const int nt0 = (wid >> 2) * 7;                // waves 0-3: nt 0..6, 4-7: nt 7..13

    v8f acc[7];
    const v8f vzero = {0.f, 0.f, 0.f, 0.f, 0.f, 0.f, 0.f, 0.f};
#pragma unroll
    for (int q = 0; q < 7; ++q) acc[q] = vzero;

    const float* imgc = img + ((size_t)b * CHANS + c) * (size_t)IMH * IMW;

    for (int kk = 0; kk < 7; ++kk) {               // 7 K-steps of 32 rows
        __syncthreads();
        // Stage 32 rows into fragment order; convert f32->f16, pack 2 halves
        // per b32 LDS store. Global reads stay fully coalesced over w.
        for (int idx = tid; idx < 16 * IMW; idx += 256) {
            int w  = idx % IMW;
            int k  = (idx / IMW) * 2;              // even k
            float a0 = imgc[(size_t)(kk * 32 + k) * IMW + w];
            float a1 = imgc[(size_t)(kk * 32 + k + 1) * IMW + w];
            int nt = w >> 4, n = w & 15;
            int lanesel = n + (k & 16);            // B layout lane
            int e = k & 15;                        // B layout element (even)
            v2h pr; pr[0] = (_Float16)a0; pr[1] = (_Float16)a1;
            *(v2h*)(&fragB[((nt * 32 + lanesel) << 4) + e]) = pr;
        }
        // Prefetch next K-slab (28 KB = 224 cachelines) while we compute.
        if (kk < 6 && tid < 224) {
            const float* nxt = imgc + (size_t)(kk + 1) * 32 * IMW;
            __builtin_prefetch(nxt + (size_t)tid * 32, 0, 0);
        }
        __syncthreads();

        // A fragment (Gaussian wy) generated in registers
        v16h a;
        {
            int   m   = lane & 15;
            int   khA = (lane >> 4) << 4;
            float y   = ys[mt * 16 + m];
#pragma unroll
            for (int e = 0; e < 16; ++e) {
                int   k = e + khA;
                float d = (float)(kk * 32 + k) - y;
                a[e]    = (_Float16)expf(-d * d * INV2S2);
            }
        }
#pragma unroll
        for (int q = 0; q < 7; ++q) {
            v16h bf = *(const v16h*)(&fragB[(((nt0 + q) * 32 + lane)) << 4]);
            acc[q]  = __builtin_amdgcn_wmma_f32_16x16x32_f16(
                          false, a, false, bf, (short)0, acc[q], false, false);
        }
    }

    // epilogue: num[s] += sum_w T(s,w) * wx(s,w)
    float tot[8];
#pragma unroll
    for (int r = 0; r < 8; ++r) tot[r] = 0.0f;
    const int wlane = lane & 15;
    const int hsel  = (lane >> 4) << 3;
#pragma unroll
    for (int q = 0; q < 7; ++q) {
        int w = (nt0 + q) * 16 + wlane;
#pragma unroll
        for (int r = 0; r < 8; ++r) {
            int   s = mt * 16 + hsel + r;
            float d = (float)w - xs[s];
            float v = acc[q][r] * expf(-d * d * INV2S2);
            v += __shfl_xor(v, 1, 32);
            v += __shfl_xor(v, 2, 32);
            v += __shfl_xor(v, 4, 32);
            v += __shfl_xor(v, 8, 32);             // stays within 16-lane half
            tot[r] += v;
        }
    }
    if (wlane == 0) {
#pragma unroll
        for (int r = 0; r < 8; ++r)
            atomicAdd(&num[mt * 16 + hsel + r], tot[r]);   // exactly 2 adds/slot
    }
    __syncthreads();

    if (tid < NS) {
        float p = num[tid] / (sy[tid] * sx[tid] + EPSV);
        pix[((size_t)b * NS + tid) * CHANS + c] = p;       // (B, n, c) flat = n*3+c
    }
}

// ---------------------------------------------------------------------------
// Kernel 2: fused MLP (6 linears) + heads + classifier (3 linears, logits).
// Single workgroup, 32 waves; activations ping-pong in LDS (f16), weights
// streamed from global directly into B fragments (128b loads).
// ---------------------------------------------------------------------------
struct MlpArgs {
    const float* W[6];  const float* bs[6];
    const float* cW[3]; const float* cb[3];
};

static __device__ void gemm_to_lds(const _Float16* __restrict__ in, int Kact,
                                   const float* __restrict__ W,
                                   const float* __restrict__ bias, int N,
                                   _Float16* __restrict__ out, bool relu, int tid) {
    const int wid = tid >> 5, lane = tid & 31;
    const int ksteps = (Kact + 31) >> 5;
    const int ntiles = (N + 15) >> 4;
    for (int t = wid; t < 2 * ntiles; t += NW2) {
        const int mt = t & 1, nt = t >> 1;
        v8f c = {0.f, 0.f, 0.f, 0.f, 0.f, 0.f, 0.f, 0.f};
        for (int ks = 0; ks < ksteps; ++ks) {
            v16h a  = ldA_lds(in + (mt * 16) * KPAD + ks * 32, KPAD, lane);
            v16h bf = ldB_wt(W, Kact, N, nt * 16, ks * 32, lane);
            c = __builtin_amdgcn_wmma_f32_16x16x32_f16(false, a, false, bf,
                                                       (short)0, c, false, false);
        }
        const int n = nt * 16 + (lane & 15);
        if (n < N) {
            const float bv = bias[n];
#pragma unroll
            for (int r = 0; r < 8; ++r) {
                int   m = mt * 16 + r + ((lane >> 4) << 3);
                float v = c[r] + bv;
                if (relu) v = fmaxf(v, 0.0f);
                out[m * KPAD + n] = (_Float16)v;
            }
        }
    }
}

__global__ __launch_bounds__(1024) void k_mlp(const float* __restrict__ pix,
                                              const float* __restrict__ loc,
                                              const float* __restrict__ scl,
                                              const float* __restrict__ hid,
                                              MlpArgs args,
                                              float* __restrict__ out_loc,
                                              float* __restrict__ out_scl,
                                              float* __restrict__ out_hid,
                                              float* __restrict__ out_logit) {
    __shared__ alignas(16) _Float16 bufA[BATCH * KPAD];   // 30720 B
    __shared__ alignas(16) _Float16 bufB[BATCH * KPAD];   // 30720 B
    const int tid = threadIdx.x, wid = tid >> 5, lane = tid & 31;

    // Phase 0: build x = [loc(2) | scale(1) | pix(192) | tanh(hidden)(256) | 0-pad]
    for (int idx = tid; idx < BATCH * KPAD; idx += 1024) {
        int m = idx / KPAD, k = idx - m * KPAD;
        float v = 0.0f;
        if (k < 2)        v = loc[m * 2 + k];
        else if (k == 2)  v = scl[m];
        else if (k < 195) v = pix[m * (NS * CHANS) + (k - 3)];
        else if (k < IN_DIM) v = tanhf(hid[m * HID + (k - 195)]);
        bufA[idx] = (_Float16)v;
    }
    __syncthreads();

    gemm_to_lds(bufA, IN_DIM, args.W[0], args.bs[0], MID, bufB, false, tid); __syncthreads();
    gemm_to_lds(bufB, MID,    args.W[1], args.bs[1], MID, bufA, true,  tid); __syncthreads();
    gemm_to_lds(bufA, MID,    args.W[2], args.bs[2], MID, bufB, true,  tid); __syncthreads();
    gemm_to_lds(bufB, MID,    args.W[3], args.bs[3], MID, bufA, true,  tid); __syncthreads();
    gemm_to_lds(bufA, MID,    args.W[4], args.bs[4], MID, bufB, true,  tid); __syncthreads();

    // Final MLP layer (N=259): heads fused into the C-fragment store.
    {
        const float* W = args.W[5]; const float* bias = args.bs[5];
        const int ntiles = (OUT_DIM + 15) >> 4;   // 17
        for (int t = wid; t < 2 * ntiles; t += NW2) {
            const int mt = t & 1, nt = t >> 1;
            v8f c = {0.f, 0.f, 0.f, 0.f, 0.f, 0.f, 0.f, 0.f};
            for (int ks = 0; ks < 8; ++ks) {
                v16h a  = ldA_lds(bufB + (mt * 16) * KPAD + ks * 32, KPAD, lane);
                v16h bf = ldB_wt(W, MID, OUT_DIM, nt * 16, ks * 32, lane);
                c = __builtin_amdgcn_wmma_f32_16x16x32_f16(false, a, false, bf,
                                                           (short)0, c, false, false);
            }
            const int n = nt * 16 + (lane & 15);
            if (n < OUT_DIM) {
                const float bv = bias[n];
#pragma unroll
                for (int r = 0; r < 8; ++r) {
                    int   m = mt * 16 + r + ((lane >> 4) << 3);
                    float v = c[r] + bv;
                    if (n < 2)       out_loc[m * 2 + n] = sigmoidf_(v);
                    else if (n == 2) out_scl[m] = sigmoidf_(v) * 0.5f;
                    else {
                        float nh = hid[m * HID + (n - 3)] + v;
                        out_hid[m * HID + (n - 3)] = nh;
                        bufA[m * KPAD + (n - 3)] = (_Float16)tanhf(nh);
                    }
                }
            }
        }
    }
    __syncthreads();

    // Classifier
    gemm_to_lds(bufA, HID, args.cW[0], args.cb[0], MID, bufB, true, tid); __syncthreads();
    gemm_to_lds(bufB, MID, args.cW[1], args.cb[1], MID, bufA, true, tid); __syncthreads();
    {
        const float* W = args.cW[2]; const float* bias = args.cb[2];
        const int ntiles = (NCLS + 15) >> 4;      // 63
        for (int t = wid; t < 2 * ntiles; t += NW2) {
            const int mt = t & 1, nt = t >> 1;
            v8f c = {0.f, 0.f, 0.f, 0.f, 0.f, 0.f, 0.f, 0.f};
            for (int ks = 0; ks < 8; ++ks) {
                v16h a  = ldA_lds(bufA + (mt * 16) * KPAD + ks * 32, KPAD, lane);
                v16h bf = ldB_wt(W, MID, NCLS, nt * 16, ks * 32, lane);
                c = __builtin_amdgcn_wmma_f32_16x16x32_f16(false, a, false, bf,
                                                           (short)0, c, false, false);
            }
            const int n = nt * 16 + (lane & 15);
            if (n < NCLS) {
                const float bv = bias[n];
#pragma unroll
                for (int r = 0; r < 8; ++r) {
                    int m = mt * 16 + r + ((lane >> 4) << 3);
                    out_logit[m * NCLS + n] = c[r] + bv;
                }
            }
        }
    }
}

// ---------------------------------------------------------------------------
// Kernel 3: in-place row softmax over (32, 1000) logits.
// ---------------------------------------------------------------------------
__global__ __launch_bounds__(256) void k_softmax(float* __restrict__ pred) {
    float* row = pred + (size_t)blockIdx.x * NCLS;
    __shared__ float red[256];
    const int tid = threadIdx.x;

    float mx = -3.4e38f;
    for (int i = tid; i < NCLS; i += 256) mx = fmaxf(mx, row[i]);
    red[tid] = mx; __syncthreads();
    for (int s = 128; s > 0; s >>= 1) {
        if (tid < s) red[tid] = fmaxf(red[tid], red[tid + s]);
        __syncthreads();
    }
    mx = red[0]; __syncthreads();

    float sm = 0.0f;
    for (int i = tid; i < NCLS; i += 256) {
        float e = expf(row[i] - mx);
        row[i] = e; sm += e;
    }
    red[tid] = sm; __syncthreads();
    for (int s = 128; s > 0; s >>= 1) {
        if (tid < s) red[tid] += red[tid + s];
        __syncthreads();
    }
    const float inv = 1.0f / red[0];
    for (int i = tid; i < NCLS; i += 256) row[i] *= inv;
}

// ---------------------------------------------------------------------------
// Launch
// d_in order (setup_inputs leaves, depth-first):
//  0 image  1 locations  2 scales  3 hidden_state
//  4..9  mlp Ws[0..5]   10..15 mlp bs[0..5]
// 16..18 cls Ws[0..2]   19..21 cls bs[0..2]
// d_out: [loc 64][scale 32][hidden 8192][pred 32000]
// ---------------------------------------------------------------------------
extern "C" void kernel_launch(void* const* d_in, const int* in_sizes, int n_in,
                              void* d_out, int out_size, void* d_ws, size_t ws_size,
                              hipStream_t stream) {
    (void)in_sizes; (void)out_size; (void)ws_size;
    if (n_in < 22) return;

    const float* image = (const float*)d_in[0];
    const float* loc   = (const float*)d_in[1];
    const float* scl   = (const float*)d_in[2];
    const float* hid   = (const float*)d_in[3];

    MlpArgs args;
    for (int i = 0; i < 6; ++i) { args.W[i]  = (const float*)d_in[4 + i];
                                  args.bs[i] = (const float*)d_in[10 + i]; }
    for (int i = 0; i < 3; ++i) { args.cW[i] = (const float*)d_in[16 + i];
                                  args.cb[i] = (const float*)d_in[19 + i]; }

    float* out      = (float*)d_out;
    float* out_loc  = out;                       // 32*2
    float* out_scl  = out + 64;                  // 32*1
    float* out_hid  = out + 96;                  // 32*256
    float* out_pred = out + 96 + BATCH * HID;    // 32*1000

    float* pix = (float*)d_ws;                   // 32*64*3 floats scratch

    k_gauss<<<dim3(BATCH * CHANS), dim3(256), 0, stream>>>(image, loc, scl, pix);
    k_mlp<<<dim3(1), dim3(1024), 0, stream>>>(pix, loc, scl, hid, args,
                                              out_loc, out_scl, out_hid, out_pred);
    k_softmax<<<dim3(BATCH), dim3(256), 0, stream>>>(out_pred);
}
